// GNNConv_89919435309312
// MI455X (gfx1250) — matrix operand order repaired and checked by reading, not compiled, not run on hardware
//
#include <hip/hip_runtime.h>

#define NN 48758          // nodes
#define DD 128            // feature dim
#define EE 750000         // edges
#define REL 5             // relations
#define EPAD (EE + REL*16)  // 750080: each relation bucket padded to x16

typedef __attribute__((ext_vector_type(16))) _Float16 v16h;
typedef __attribute__((ext_vector_type(8)))  _Float16 v8h;
typedef __attribute__((ext_vector_type(8)))  float    v8f;

// ---------------- small helpers ----------------

__device__ __forceinline__ unsigned fenc(float f) {
  // monotone float->uint mapping so atomicMax(u32) == float max
  unsigned u = __float_as_uint(f);
  return (u & 0x80000000u) ? ~u : (u | 0x80000000u);
}
__device__ __forceinline__ float fdec(unsigned u) {
  return (u & 0x80000000u) ? __uint_as_float(u & 0x7FFFFFFFu) : __uint_as_float(~u);
}

__global__ void fill_u32(unsigned* p, unsigned v, int n) {
  int gid = blockIdx.x * blockDim.x + threadIdx.x;
  if (gid < n) p[gid] = v;
}

__global__ void cvt_f16(const float* in, _Float16* out, int n) {
  int gid = blockIdx.x * blockDim.x + threadIdx.x;
  if (gid < n) out[gid] = (_Float16)in[gid];
}

// Pack a 128x128 row-major f32 weight (W[k][n]) into B-fragment order:
// P[((kt*8+nt)*32+lane)*16+h] = W[k][n],  n = nt*16 + (lane&15),
// k = kt*32 + ((lane>>4)&1)*8 + (h&7) + (h>>3)*16   -> inner-loop B load is one contiguous v16h.
__global__ void pack_w16(const float* __restrict__ W, _Float16* __restrict__ P) {
  int t = blockIdx.x * blockDim.x + threadIdx.x;   // 0..16383
  int h    = t & 15;
  int lane = (t >> 4) & 31;
  int nt   = (t >> 9) & 7;
  int kt   = (t >> 12) & 3;
  int n = nt * 16 + (lane & 15);
  int k = kt * 32 + ((lane >> 4) & 1) * 8 + (h & 7) + (h >> 3) * 16;
  P[t] = (_Float16)W[k * DD + n];
}

// Load one A fragment (16x32 f16, ISA layout) for this lane.
// p16 points at row_base + kt*32 in a row-major [*,128] f16 matrix.
__device__ __forceinline__ v16h load_a_frag(const _Float16* p16, int lane) {
  const v8h* p = (const v8h*)(p16 + ((lane >> 4) & 1) * 8);
  v8h lo = p[0];   // K = off+0..7   (pairs -> VGPR0..3)
  v8h hi = p[2];   // K = off+16..23 (pairs -> VGPR4..7)
  v16h a;
#pragma unroll
  for (int i = 0; i < 8; ++i) { a[i] = lo[i]; a[i + 8] = hi[i]; }
  return a;
}

// cooperative copy of one packed 128x128 f16 weight (32KB) into LDS
__device__ __forceinline__ void stage_b_lds(_Float16* Bsh, const _Float16* Wp, int tid) {
  const v16h* g = (const v16h*)Wp;
  v16h* l = (v16h*)Bsh;
#pragma unroll
  for (int i = 0; i < 4; ++i) l[tid + i * 256] = g[tid + i * 256];  // 1024 chunks / 256 thr
}

// one k-step of the 16x128x128 wave tile: prefetch all 8 B fragments, then 8 WMMAs
__device__ __forceinline__ void mac_step_lds(const _Float16* Bsh, int kt, int lane,
                                             v16h a, v8f acc[8]) {
  v16h bf[8];
#pragma unroll
  for (int nt = 0; nt < 8; ++nt)
    bf[nt] = *(const v16h*)(Bsh + (((kt * 8 + nt) * 32 + lane) << 4));
#pragma unroll
  for (int nt = 0; nt < 8; ++nt)
    acc[nt] = __builtin_amdgcn_wmma_f32_16x16x32_f16(false, a, false, bf[nt],
                                                     (short)0, acc[nt], false, false);
}

__device__ __forceinline__ void mac_step_glb(const _Float16* Wp, int kt, int lane,
                                             v16h a, v8f acc[8]) {
  v16h bf[8];
#pragma unroll
  for (int nt = 0; nt < 8; ++nt)
    bf[nt] = *(const v16h*)(Wp + (((kt * 8 + nt) * 32 + lane) << 4));
#pragma unroll
  for (int nt = 0; nt < 8; ++nt)
    acc[nt] = __builtin_amdgcn_wmma_f32_16x16x32_f16(false, a, false, bf[nt],
                                                     (short)0, acc[nt], false, false);
}

// ---------------- dense node GEMM: C[N,128] = A16[N,128] @ Wp + bias ----------------
// 256 threads = 8 waves; B staged in LDS once per block; each wave owns a 16-row tile.
__global__ void __launch_bounds__(256) gemm_node(
    const _Float16* __restrict__ A16, const _Float16* __restrict__ Wp,
    const float* __restrict__ bias,
    float* __restrict__ C32, _Float16* __restrict__ C16, int nrows) {
  __shared__ _Float16 Bsh[DD * DD];   // 32KB
  int tid = threadIdx.x;
  int lane = tid & 31;
  int w = tid >> 5;

  stage_b_lds(Bsh, Wp, tid);
  __syncthreads();

  int row0 = blockIdx.x * 128 + w * 16;
  int row = row0 + (lane & 15);
  int rsafe = (row < nrows) ? row : 0;
  const _Float16* arow = A16 + (size_t)rsafe * DD;

  v8f acc[8];
#pragma unroll
  for (int i = 0; i < 8; ++i)
#pragma unroll
    for (int j = 0; j < 8; ++j) acc[i][j] = 0.0f;

#pragma unroll
  for (int kt = 0; kt < 4; ++kt) {
    v16h a = load_a_frag(arow + kt * 32, lane);
    mac_step_lds(Bsh, kt, lane, a, acc);
  }
  // C layout: lane -> col n=lane&15 ; VGPR j -> row (lane>=16 ? 8 : 0)+j
  int nlo = lane & 15;
  int rbase = row0 + ((lane >> 4) << 3);
#pragma unroll
  for (int nt = 0; nt < 8; ++nt) {
    float bv = bias[nt * 16 + nlo];
#pragma unroll
    for (int j = 0; j < 8; ++j) {
      int r = rbase + j;
      if (r < nrows) {
        float v = acc[nt][j] + bv;
        C32[(size_t)r * DD + nt * 16 + nlo] = v;
        C16[(size_t)r * DD + nt * 16 + nlo] = (_Float16)v;
      }
    }
  }
}

// ---------------- per-edge relation GEMM ----------------
// 8 edge-tiles (128 edges) per 256-thread block from the relation-bucketed perm[].
// Common case: whole block is one relation -> B staged in LDS. Boundary blocks
// (at most REL-1 in the grid) fall back to per-wave global B loads.
__global__ void __launch_bounds__(256) gemm_edge(
    const _Float16* __restrict__ xlin16, const _Float16* __restrict__ WpAll,
    const int* __restrict__ perm, const int* __restrict__ relStart,
    const int* __restrict__ src, _Float16* __restrict__ res16) {
  __shared__ _Float16 Bsh[DD * DD];   // 32KB
  int tid = threadIdx.x;
  int lane = tid & 31;
  int w = tid >> 5;
  int blockBase = blockIdx.x * 128;
  int base = blockBase + w * 16;

  // block-uniform relation range (relStart is the same for every thread)
  int rFirst = 0, rLast = 0, r = 0;
#pragma unroll
  for (int i = 1; i < REL; ++i) {
    rFirst = (blockBase >= relStart[i]) ? i : rFirst;
    rLast  = (blockBase + 112 >= relStart[i]) ? i : rLast;
    r      = (base >= relStart[i]) ? i : r;
  }
  bool uniformRel = (rFirst == rLast);

  int e = perm[base + (lane & 15)];
  int srow = (e >= 0) ? src[e] : 0;
  const _Float16* arow = xlin16 + (size_t)srow * DD;

  v8f acc[8];
#pragma unroll
  for (int i = 0; i < 8; ++i)
#pragma unroll
    for (int j = 0; j < 8; ++j) acc[i][j] = 0.0f;

  if (uniformRel) {
    stage_b_lds(Bsh, WpAll + (size_t)rFirst * (DD * DD), tid);
    __syncthreads();
#pragma unroll
    for (int kt = 0; kt < 4; ++kt) {
      v16h a;
      if (e >= 0) {
        a = load_a_frag(arow + kt * 32, lane);
      } else {
#pragma unroll
        for (int i = 0; i < 16; ++i) a[i] = (_Float16)0.0f;
      }
      mac_step_lds(Bsh, kt, lane, a, acc);
    }
  } else {
    const _Float16* Wp = WpAll + (size_t)r * (DD * DD);
#pragma unroll
    for (int kt = 0; kt < 4; ++kt) {
      v16h a;
      if (e >= 0) {
        a = load_a_frag(arow + kt * 32, lane);
      } else {
#pragma unroll
        for (int i = 0; i < 16; ++i) a[i] = (_Float16)0.0f;
      }
      mac_step_glb(Wp, kt, lane, a, acc);
    }
  }

  int nlo = lane & 15;
  int mbase = (lane >> 4) << 3;
#pragma unroll
  for (int j = 0; j < 8; ++j) {
    int eM = perm[base + mbase + j];
    if (eM >= 0) {
      _Float16* dstp = res16 + (size_t)eM * DD + nlo;
#pragma unroll
      for (int nt = 0; nt < 8; ++nt) dstp[nt * 16] = (_Float16)acc[nt][j];
    }
  }
}

// ---------------- batchnorm ----------------
__global__ void bn_stats(const float* __restrict__ h, float* sum, float* sq) {
  int c = threadIdx.x;       // 128 threads = channels
  float a = 0.0f, a2 = 0.0f;
  for (int row = blockIdx.x; row < NN; row += gridDim.x) {
    float v = h[(size_t)row * DD + c];
    a += v; a2 += v * v;
  }
  atomicAdd(&sum[c], a);
  atomicAdd(&sq[c], a2);
}

__global__ void bn_apply(float* __restrict__ h32, _Float16* __restrict__ h16,
                         const float* __restrict__ sum, const float* __restrict__ sq,
                         const float* __restrict__ g, const float* __restrict__ b) {
  int gid = blockIdx.x * blockDim.x + threadIdx.x;
  if (gid >= NN * DD) return;
  int c = gid & (DD - 1);
  float mu = sum[c] * (1.0f / NN);
  float var = sq[c] * (1.0f / NN) - mu * mu;
  float v = (h32[gid] - mu) * rsqrtf(var + 1e-5f) * g[c] + b[c];
  v = v > 0.0f ? v : 0.0f;
  h32[gid] = v;
  h16[gid] = (_Float16)v;
}

// ---------------- graph prep ----------------
__global__ void deg_count(const int* __restrict__ src, float* deg) {
  int e = blockIdx.x * blockDim.x + threadIdx.x;
  if (e < EE) atomicAdd(&deg[src[e]], 1.0f);
}
__global__ void make_dis(const float* __restrict__ deg, float* dis) {
  int i = blockIdx.x * blockDim.x + threadIdx.x;
  if (i < NN) { float d = deg[i]; dis[i] = d > 0.0f ? rsqrtf(d) : 0.0f; }
}
__global__ void rel_count(const int* __restrict__ et, int* cnt) {
  int e = blockIdx.x * blockDim.x + threadIdx.x;
  if (e < EE) atomicAdd(&cnt[et[e]], 1);
}
__global__ void rel_offsets(const int* __restrict__ cnt, int* start) {
  int s = 0;
  for (int r = 0; r < REL; ++r) { start[r] = s; s += ((cnt[r] + 15) / 16) * 16; }
  start[REL] = s;
}
__global__ void rel_scatter(const int* __restrict__ et, const int* __restrict__ start,
                            int* cursor, int* perm) {
  int e = blockIdx.x * blockDim.x + threadIdx.x;
  if (e < EE) {
    int r = et[e];
    int pos = atomicAdd(&cursor[r], 1);
    perm[start[r] + pos] = e;
  }
}

// ---------------- scatter passes (thread per edge x channel) ----------------
__global__ void scatter_gcn(const float* __restrict__ xlin32, const int* __restrict__ dst,
                            const int* __restrict__ src, const float* __restrict__ dis,
                            float* msgg) {
  int gid = blockIdx.x * blockDim.x + threadIdx.x;
  if (gid >= EE * DD) return;
  int e = gid >> 7, c = gid & (DD - 1);
  int d = dst[e], s = src[e];
  float nrm = dis[d] * dis[s];
  atomicAdd(&msgg[(size_t)d * DD + c], xlin32[(size_t)s * DD + c] * nrm);
}
__global__ void scatter_max(const _Float16* __restrict__ res16, const int* __restrict__ dst,
                            unsigned* menc) {
  int gid = blockIdx.x * blockDim.x + threadIdx.x;
  if (gid >= EE * DD) return;
  int e = gid >> 7, c = gid & (DD - 1);
  atomicMax(&menc[(size_t)dst[e] * DD + c], fenc((float)res16[gid]));
}
__global__ void scatter_expsum(const _Float16* __restrict__ res16, const int* __restrict__ dst,
                               const unsigned* __restrict__ menc, float* sbuf) {
  int gid = blockIdx.x * blockDim.x + threadIdx.x;
  if (gid >= EE * DD) return;
  int e = gid >> 7, c = gid & (DD - 1);
  size_t di = (size_t)dst[e] * DD + c;
  atomicAdd(&sbuf[di], __expf((float)res16[gid] - fdec(menc[di])));
}
__global__ void scatter_wsum(const _Float16* __restrict__ res16, const int* __restrict__ dst,
                             const unsigned* __restrict__ menc, const float* __restrict__ sbuf,
                             float* msg) {
  int gid = blockIdx.x * blockDim.x + threadIdx.x;
  if (gid >= EE * DD) return;
  int e = gid >> 7, c = gid & (DD - 1);
  size_t di = (size_t)dst[e] * DD + c;
  float v = (float)res16[gid];
  float w = __expf(v - fdec(menc[di])) / sbuf[di];
  atomicAdd(&msg[di], v * w);
}

__global__ void combine(const float* __restrict__ msgg, const float* __restrict__ msg,
                        float* t32, _Float16* t16) {
  int gid = blockIdx.x * blockDim.x + threadIdx.x;
  if (gid >= NN * DD) return;
  float v = msgg[gid] + 0.1f * fmaxf(msg[gid], 0.0f);
  t32[gid] = v;
  t16[gid] = (_Float16)v;
}

__global__ void gelu_gather(const float* __restrict__ h32, const int* __restrict__ idx,
                            float* out) {
  int gid = blockIdx.x * blockDim.x + threadIdx.x;
  if (gid >= 4096 * DD) return;
  int i = gid >> 7, c = gid & (DD - 1);
  float x = h32[(size_t)idx[i] * DD + c];
  out[gid] = 0.5f * x * (1.0f + erff(x * 0.70710678118654752f));
}

// ---------------- launcher ----------------
extern "C" void kernel_launch(void* const* d_in, const int* in_sizes, int n_in,
                              void* d_out, int out_size, void* d_ws, size_t ws_size,
                              hipStream_t stream) {
  const float* x      = (const float*)d_in[0];
  const int*   ei     = (const int*)d_in[1];
  const int*   et     = (const int*)d_in[2];
  const int*   idx    = (const int*)d_in[4];
  const float* proj_w = (const float*)d_in[5];
  const float* proj_b = (const float*)d_in[6];
  const float* bn_g   = (const float*)d_in[7];
  const float* bn_b   = (const float*)d_in[8];
  const float* win_w[2]  = {(const float*)d_in[9],  (const float*)d_in[14]};
  const float* win_b[2]  = {(const float*)d_in[10], (const float*)d_in[15]};
  const float* wrelw[2]  = {(const float*)d_in[11], (const float*)d_in[16]};
  const float* cout_w[2] = {(const float*)d_in[12], (const float*)d_in[17]};
  const float* cout_b[2] = {(const float*)d_in[13], (const float*)d_in[18]};
  float* out = (float*)d_out;
  const int* dstp = ei;
  const int* srcp = ei + EE;

  // workspace carve-out
  char* ws = (char*)d_ws;
  size_t off = 0;
  auto take = [&](size_t bytes) -> char* {
    char* p = ws + off;
    off = (off + bytes + 255) & ~(size_t)255;
    return p;
  };
  _Float16* packs = (_Float16*)take((size_t)15 * DD * DD * 2);  // 15 packed weights
  _Float16* x16   = (_Float16*)take((size_t)NN * DD * 2);
  float*    h32   = (float*)take((size_t)NN * DD * 4);
  _Float16* h16   = (_Float16*)take((size_t)NN * DD * 2);
  float*    t32   = (float*)take((size_t)NN * DD * 4);
  _Float16* t16   = (_Float16*)take((size_t)NN * DD * 2);
  float*    bnsum = (float*)take(DD * 4);
  float*    bnsq  = (float*)take(DD * 4);
  float*    deg   = (float*)take((size_t)NN * 4);
  float*    dis   = (float*)take((size_t)NN * 4);
  int* relCount   = (int*)take(REL * 4);
  int* relStart   = (int*)take((REL + 1) * 4);
  int* cursor     = (int*)take(REL * 4);
  int* perm       = (int*)take((size_t)EPAD * 4);
  _Float16* res16 = (_Float16*)take((size_t)EE * DD * 2);
  unsigned* menc  = (unsigned*)take((size_t)NN * DD * 4);
  float* sbuf     = (float*)take((size_t)NN * DD * 4);
  float* msg      = (float*)take((size_t)NN * DD * 4);
  float* msgg     = (float*)take((size_t)NN * DD * 4);

  const int ND = NN * DD;            // 6,241,024
  const int ED = EE * DD;            // 96,000,000
  auto FB = [](int n) { return (n + 255) / 256; };
  const int gemmNodeGrid = (NN + 127) / 128;   // 381 blocks x 8 waves
  const int gemmEdgeGrid = EPAD / 128;         // 5860 blocks x 8 waves

  // 1) pack all weights to f16 fragment layout
  pack_w16<<<64, 256, 0, stream>>>(proj_w, packs + 0 * 16384);
  pack_w16<<<64, 256, 0, stream>>>(win_w[0], packs + 1 * 16384);
  for (int r = 0; r < REL; ++r)
    pack_w16<<<64, 256, 0, stream>>>(wrelw[0] + (size_t)r * 16384, packs + (2 + r) * 16384);
  pack_w16<<<64, 256, 0, stream>>>(cout_w[0], packs + 7 * 16384);
  pack_w16<<<64, 256, 0, stream>>>(win_w[1], packs + 8 * 16384);
  for (int r = 0; r < REL; ++r)
    pack_w16<<<64, 256, 0, stream>>>(wrelw[1] + (size_t)r * 16384, packs + (9 + r) * 16384);
  pack_w16<<<64, 256, 0, stream>>>(cout_w[1], packs + 14 * 16384);

  // 2) input proj + BN + relu
  cvt_f16<<<FB(ND), 256, 0, stream>>>(x, x16, ND);
  gemm_node<<<gemmNodeGrid, 256, 0, stream>>>(x16, packs + 0 * 16384, proj_b, h32, h16, NN);
  fill_u32<<<1, 256, 0, stream>>>((unsigned*)bnsum, 0u, DD);
  fill_u32<<<1, 256, 0, stream>>>((unsigned*)bnsq, 0u, DD);
  bn_stats<<<512, 128, 0, stream>>>(h32, bnsum, bnsq);
  bn_apply<<<FB(ND), 256, 0, stream>>>(h32, h16, bnsum, bnsq, bn_g, bn_b);

  // 3) gcn-norm degrees (layer-invariant)
  fill_u32<<<FB(NN), 256, 0, stream>>>((unsigned*)deg, 0u, NN);
  deg_count<<<FB(EE), 256, 0, stream>>>(srcp, deg);
  make_dis<<<FB(NN), 256, 0, stream>>>(deg, dis);

  // 4) relation buckets, padded to multiples of 16 (layer-invariant)
  fill_u32<<<1, 256, 0, stream>>>((unsigned*)relCount, 0u, REL);
  fill_u32<<<1, 256, 0, stream>>>((unsigned*)cursor, 0u, REL);
  rel_count<<<FB(EE), 256, 0, stream>>>(et, relCount);
  rel_offsets<<<1, 1, 0, stream>>>(relCount, relStart);
  fill_u32<<<FB(EPAD), 256, 0, stream>>>((unsigned*)perm, 0xFFFFFFFFu, EPAD);
  rel_scatter<<<FB(EE), 256, 0, stream>>>(et, relStart, cursor, perm);

  // 5) two DAN layers
  for (int L = 0; L < 2; ++L) {
    const _Float16* winP  = packs + (L ? 8 : 1) * 16384;
    const _Float16* wrelP = packs + (L ? 9 : 2) * 16384;
    const _Float16* coutP = packs + (L ? 14 : 7) * 16384;

    // xlin = h @ win + b  (t32/t16 hold xlin)
    gemm_node<<<gemmNodeGrid, 256, 0, stream>>>(h16, winP, win_b[L], t32, t16, NN);

    // gcn-norm scatter sum
    fill_u32<<<FB(ND), 256, 0, stream>>>((unsigned*)msgg, 0u, ND);
    scatter_gcn<<<FB(ED), 256, 0, stream>>>(t32, dstp, srcp, dis, msgg);

    // per-edge relation GEMM -> res16  (the 49 GFLOP WMMA workhorse)
    gemm_edge<<<gemmEdgeGrid, 256, 0, stream>>>(t16, wrelP, perm, relStart, srcp, res16);

    // scatter-softmax over destination neighborhoods
    fill_u32<<<FB(ND), 256, 0, stream>>>(menc, 0u, ND);
    scatter_max<<<FB(ED), 256, 0, stream>>>(res16, dstp, menc);
    fill_u32<<<FB(ND), 256, 0, stream>>>((unsigned*)sbuf, 0u, ND);
    scatter_expsum<<<FB(ED), 256, 0, stream>>>(res16, dstp, menc, sbuf);
    fill_u32<<<FB(ND), 256, 0, stream>>>((unsigned*)msg, 0u, ND);
    scatter_wsum<<<FB(ED), 256, 0, stream>>>(res16, dstp, menc, sbuf, msg);

    // combine (reuse t32/t16) then output GEMM back into h32/h16
    combine<<<FB(ND), 256, 0, stream>>>(msgg, msg, t32, t16);
    gemm_node<<<gemmNodeGrid, 256, 0, stream>>>(t16, coutP, cout_b[L], h32, h16, NN);
  }

  // 6) exact GELU + index gather
  gelu_gather<<<FB(4096 * DD), 256, 0, stream>>>(h32, idx, out);
}
